// PraxisNano_60919816127055
// MI455X (gfx1250) — matrix operand order repaired and checked by reading, not compiled, not run on hardware
//
#include <hip/hip_runtime.h>
#include <stdint.h>

// ---------------------------------------------------------------------------
// Chunked FFT-mix + MLP scan for MI455X (gfx1250, wave32, WMMA).
// Compute-bound (~0.83 TFLOP vs ~8us of HBM traffic at 23.3 TB/s): both big
// GEMMs run on v_wmma_f32_16x16x32_bf16 (fp32 accumulate) with:
//   - double-buffered LDS tiles, one barrier per K-step
//   - GLOBAL_LOAD_ASYNC_TO_LDS_B128 staging (ASYNCcnt) so tile copies land
//     directly in LDS and overlap the WMMA pipe (no VGPR round trip)
//   - split-K x4 on the down-projection for occupancy (it sits on the
//     128-step serial critical path)
// ---------------------------------------------------------------------------

#define E_     1024
#define DFF_   4096
#define B_     4
#define T_     4096
#define NCHUNK 128
#define ROWS   (B_ * 64)   // 256 rows processed per step
#define LDT    40          // padded LDS tile row stride (halves); 80B rows
#define KSPLIT 4           // split-K factor for the down GEMM

typedef __attribute__((ext_vector_type(16))) __bf16 v16bf;
typedef __attribute__((ext_vector_type(8)))  float  v8f;

union FragU { v16bf v; uint4 q[2]; };

__device__ __forceinline__ unsigned short f2bf(float f) {
    union { float f; unsigned u; } x; x.f = f;
    unsigned r = x.u + 0x7FFFu + ((x.u >> 16) & 1u);   // round-to-nearest-even
    return (unsigned short)(r >> 16);
}

// --- CDNA5 async memory->LDS staging (tracked by ASYNCcnt) -----------------
// GVS form: 64-bit SGPR base + per-lane 32-bit byte offset; %0 = LDS byte
// address VGPR (used as is, no offset).
__device__ __forceinline__ void async_b128(unsigned lds, unsigned voff,
                                           unsigned long long base) {
    asm volatile("global_load_async_to_lds_b128 %0, %1, %2"
                 :: "v"(lds), "v"(voff), "s"(base) : "memory");
}
__device__ __forceinline__ void wait_async_all() {
    asm volatile("s_wait_asynccnt 0x0" ::: "memory");
}
__device__ __forceinline__ unsigned lds_addr(const void* p) {
    return (unsigned)(uintptr_t)p;   // low 32 bits of generic shared ptr = LDS addr
}

// ---------------------------------------------------------------------------
// One-time prep kernels
// ---------------------------------------------------------------------------

__global__ void k_cvt(const float* __restrict__ wup, const float* __restrict__ wdn,
                      unsigned short* __restrict__ upb, unsigned short* __restrict__ dnb) {
    const long n1 = (long)2 * DFF_ * E_;
    const long n2 = (long)E_ * DFF_;
    for (long i = blockIdx.x * (long)blockDim.x + threadIdx.x; i < n1 + n2;
         i += (long)gridDim.x * blockDim.x) {
        if (i < n1) upb[i] = f2bf(wup[i]);
        else        dnb[i - n1] = f2bf(wdn[i - n1]);
    }
}

// M[t][tp] = sum_u w_fft2[t,u] * w_fft1[u,tp]  (64x64), plus row sums of M.
__global__ void k_mmat(const float* __restrict__ w1, const float* __restrict__ w2,
                       float* __restrict__ Mmat, float* __restrict__ rowsum) {
    __shared__ float Ms[64 * 64];
    for (int r = 0; r < 16; ++r) {
        int lin = threadIdx.x + r * 256;
        int t = lin >> 6, tp = lin & 63;
        float a = 0.f;
        for (int u = 0; u < 48; ++u) a += w2[t * 48 + u] * w1[u * 64 + tp];
        Ms[lin] = a;
        Mmat[lin] = a;
    }
    __syncthreads();
    if (threadIdx.x < 64) {
        float s = 0.f;
        for (int tp = 0; tp < 64; ++tp) s += Ms[threadIdx.x * 64 + tp];
        rowsum[threadIdx.x] = s;
    }
}

// init_prev = concat(zeros(B,32,E), x[:, :32, :])
__global__ void k_init(const float* __restrict__ x, float* __restrict__ prev) {
    long n = (long)ROWS * E_;
    for (long i = blockIdx.x * (long)blockDim.x + threadIdx.x; i < n;
         i += (long)gridDim.x * blockDim.x) {
        int e = (int)(i % E_);
        int row = (int)(i / E_);
        int b = row >> 6, t = row & 63;
        prev[i] = (t < 32) ? 0.f : x[((size_t)b * T_ + (t - 32)) * E_ + e];
    }
}

// ---------------------------------------------------------------------------
// Per-step kernels
// ---------------------------------------------------------------------------

// LN1 row stats over "mixed" rows (prev proc rows 32:64 / fresh x rows).
__global__ void k_stats1(const float* __restrict__ prevp, const float* __restrict__ x,
                         int chunk, float* __restrict__ mu, float* __restrict__ rs) {
    int row = blockIdx.x;            // 0..255
    int b = row >> 6, t = row & 63;
    float s = 0.f, s2 = 0.f;
    for (int e = threadIdx.x; e < E_; e += 256) {
        float v;
        if (t < 32) v = prevp[(size_t)(b * 64 + t + 32) * E_ + e];
        else {
            int gt = chunk * 32 + t;
            v = (gt < T_) ? x[((size_t)b * T_ + gt) * E_ + e] : 0.f;
        }
        s += v; s2 += v * v;
    }
    __shared__ float sh[256], sh2[256];
    sh[threadIdx.x] = s; sh2[threadIdx.x] = s2;
    __syncthreads();
    for (int o = 128; o > 0; o >>= 1) {
        if (threadIdx.x < o) { sh[threadIdx.x] += sh[threadIdx.x + o];
                               sh2[threadIdx.x] += sh2[threadIdx.x + o]; }
        __syncthreads();
    }
    if (threadIdx.x == 0) {
        float m = sh[0] / (float)E_;
        float var = sh2[0] / (float)E_ - m * m;
        mu[row] = m; rs[row] = rsqrtf(var + 1e-5f);
    }
}

// res2[b,t,e] = ln1_w[e]*sum_tp M[t,tp]*(mixed[b,tp,e]-mu)*rs + ln1_b[e]*rowsumM[t] + x_pad
__global__ void k_mix(const float* __restrict__ prevp, const float* __restrict__ x,
                      const float* __restrict__ Mmat, const float* __restrict__ rowsum,
                      const float* __restrict__ mu1, const float* __restrict__ rs1,
                      const float* __restrict__ ln1w, const float* __restrict__ ln1b,
                      int chunk, float* __restrict__ res2) {
    int b  = blockIdx.x >> 4;
    int ec = (blockIdx.x & 15) * 64;
    __shared__ float Ml[64][64];
    __shared__ float cn[64][65];
    for (int r = 0; r < 16; ++r) {
        int lin = threadIdx.x + r * 256;
        Ml[lin >> 6][lin & 63] = Mmat[lin];
    }
    for (int r = 0; r < 16; ++r) {
        int lin = threadIdx.x + r * 256;
        int tp = lin >> 6, el = lin & 63, e = ec + el;
        float v;
        if (tp < 32) v = prevp[(size_t)(b * 64 + tp + 32) * E_ + e];
        else {
            int gt = chunk * 32 + tp;
            v = (gt < T_) ? x[((size_t)b * T_ + gt) * E_ + e] : 0.f;
        }
        int row = b * 64 + tp;
        cn[tp][el] = (v - mu1[row]) * rs1[row];
    }
    __syncthreads();
    for (int r = 0; r < 16; ++r) {
        int lin = threadIdx.x + r * 256;
        int t = lin >> 6, el = lin & 63, e = ec + el;
        float acc = 0.f;
#pragma unroll 8
        for (int tp = 0; tp < 64; ++tp) acc += Ml[t][tp] * cn[tp][el];
        int gt = chunk * 32 + t;
        float resid = (gt < T_) ? x[((size_t)b * T_ + gt) * E_ + e] : 0.f;
        res2[(size_t)(b * 64 + t) * E_ + e] = ln1w[e] * acc + ln1b[e] * rowsum[t] + resid;
    }
}

// Fused LN2: row stats + normalize + bf16 write of cn2. One block per row.
__global__ void k_norm2(const float* __restrict__ res2,
                        const float* __restrict__ ln2w, const float* __restrict__ ln2b,
                        unsigned short* __restrict__ cn2) {
    int row = blockIdx.x;
    float s = 0.f, s2 = 0.f;
    for (int e = threadIdx.x; e < E_; e += 256) {
        float v = res2[(size_t)row * E_ + e];
        s += v; s2 += v * v;
    }
    __shared__ float sh[256], sh2[256];
    __shared__ float sm, sr;
    sh[threadIdx.x] = s; sh2[threadIdx.x] = s2;
    __syncthreads();
    for (int o = 128; o > 0; o >>= 1) {
        if (threadIdx.x < o) { sh[threadIdx.x] += sh[threadIdx.x + o];
                               sh2[threadIdx.x] += sh2[threadIdx.x + o]; }
        __syncthreads();
    }
    if (threadIdx.x == 0) {
        float m = sh[0] / (float)E_;
        float var = sh2[0] / (float)E_ - m * m;
        sm = m; sr = rsqrtf(var + 1e-5f);
    }
    __syncthreads();
    float m = sm, r = sr;
    for (int e = threadIdx.x; e < E_; e += 256) {
        float v = res2[(size_t)row * E_ + e];
        cn2[(size_t)row * E_ + e] = f2bf((v - m) * r * ln2w[e] + ln2b[e]);
    }
}

// ---------------------------------------------------------------------------
// WMMA fragment loaders (wave32, v_wmma_f32_16x16x32_bf16 layouts per ISA):
//  A 16x32:  lane<16 -> halves K0..7 / K16..23 ; lane>=16 -> K8..15 / K24..31
//  B 32x16:  N = lane&15 ; lane<16 -> K0..7 / K16..23 ; lane>=16 -> K8..15 / K24..31  (B tile staged N-major in LDS)
//  C 16x16:  M = v + 8*(lane>>4) ; N = lane&15
// ---------------------------------------------------------------------------

__device__ __forceinline__ v16bf load_a_frag(const unsigned short* As, int rowBase, int lane) {
    FragU f;
    int rr = rowBase + (lane & 15);
    int ks = (lane >> 4) * 8;
    f.q[0] = *(const uint4*)&As[rr * LDT + ks];
    f.q[1] = *(const uint4*)&As[rr * LDT + ks + 16];
    return f.v;
}
__device__ __forceinline__ v16bf load_b_frag(const unsigned short* Bs, int nBase, int lane) {
    FragU f;
    int nr = nBase + (lane & 15);
    int ks = (lane >> 4) * 8;
    f.q[0] = *(const uint4*)&Bs[nr * LDT + ks];
    f.q[1] = *(const uint4*)&Bs[nr * LDT + ks + 16];
    return f.v;
}

// gv = cn2 @ w_up^T (+b_up); f = __sinf(gate)*val -> bf16
// grid (4096/128, 256/64); block tile 64 rows x (128 gate + 128 val) cols.
// Double-buffered LDS fed by async-to-LDS loads; one barrier per K-step.
__global__ __launch_bounds__(256) void k_mlp_up(
    const unsigned short* __restrict__ cn2, const unsigned short* __restrict__ upb,
    const float* __restrict__ bup, unsigned short* __restrict__ fbf) {
    const int tid = threadIdx.x;
    const int wave = tid >> 5, lane = tid & 31;
    const int wm = wave & 1, wn = wave >> 1;        // 2 x 4 wave grid
    const int m0 = blockIdx.y * 64;
    const int f0 = blockIdx.x * 128;

    __shared__ __align__(16) unsigned short As[2][64 * LDT];
    __shared__ __align__(16) unsigned short Bg[2][128 * LDT];
    __shared__ __align__(16) unsigned short Bv[2][128 * LDT];

    v8f accg[2][2] = {}; v8f accv[2][2] = {};

    // staging coordinates (byte offsets for async loads)
    const int arow = tid >> 2, akof = (tid & 3) * 8;     // A: 64r x 4 chunks of 8 halves
    const int brow = tid >> 1, bkof = (tid & 1) * 16;    // B: 128r x 2 chunks of 16 halves
    const unsigned aByte = (unsigned)(((size_t)(m0 + arow) * E_ + akof) * 2);
    const unsigned gByte = (unsigned)(((size_t)(f0 + brow) * E_ + bkof) * 2);
    const unsigned vByte = (unsigned)(((size_t)(f0 + DFF_ + brow) * E_ + bkof) * 2);
    const int aL = arow * LDT + akof;
    const int bL = brow * LDT + bkof;
    const unsigned ldsA[2] = { lds_addr(&As[0][aL]), lds_addr(&As[1][aL]) };
    const unsigned ldsG[2] = { lds_addr(&Bg[0][bL]), lds_addr(&Bg[1][bL]) };
    const unsigned ldsV[2] = { lds_addr(&Bv[0][bL]), lds_addr(&Bv[1][bL]) };
    const unsigned long long cn2b = (unsigned long long)(uintptr_t)cn2;
    const unsigned long long upbb = (unsigned long long)(uintptr_t)upb;

    auto stage = [&](int buf, int k) {   // async: memory -> LDS buffer `buf`
        unsigned kb = (unsigned)k * 2;
        async_b128(ldsA[buf],      aByte + kb,      cn2b);
        async_b128(ldsG[buf],      gByte + kb,      upbb);
        async_b128(ldsG[buf] + 16, gByte + kb + 16, upbb);
        async_b128(ldsV[buf],      vByte + kb,      upbb);
        async_b128(ldsV[buf] + 16, vByte + kb + 16, upbb);
    };

    stage(0, 0);
    wait_async_all();
    __syncthreads();

    auto step = [&](int cur, int nxt, int kn) {
        if (kn < E_) stage(nxt, kn);     // overlaps the WMMA block below
        v16bf a[2], bg[2], bv[2];
#pragma unroll
        for (int mi = 0; mi < 2; ++mi) a[mi] = load_a_frag(As[cur], wm * 32 + mi * 16, lane);
#pragma unroll
        for (int ni = 0; ni < 2; ++ni) {
            bg[ni] = load_b_frag(Bg[cur], wn * 32 + ni * 16, lane);
            bv[ni] = load_b_frag(Bv[cur], wn * 32 + ni * 16, lane);
        }
#pragma unroll
        for (int mi = 0; mi < 2; ++mi)
#pragma unroll
            for (int ni = 0; ni < 2; ++ni) {
                accg[mi][ni] = __builtin_amdgcn_wmma_f32_16x16x32_bf16(
                    false, a[mi], false, bg[ni], (short)0, accg[mi][ni], false, false);
                accv[mi][ni] = __builtin_amdgcn_wmma_f32_16x16x32_bf16(
                    false, a[mi], false, bv[ni], (short)0, accv[mi][ni], false, false);
            }
        wait_async_all();
        __syncthreads();
    };

    for (int k0 = 0; k0 < E_; k0 += 64) {
        step(0, 1, k0 + 32);
        step(1, 0, k0 + 64);
    }

    const int ch = (lane >> 4) * 8, cn_ = lane & 15;
#pragma unroll
    for (int mi = 0; mi < 2; ++mi)
#pragma unroll
        for (int ni = 0; ni < 2; ++ni)
#pragma unroll
            for (int v = 0; v < 8; ++v) {
                int gr = m0 + wm * 32 + mi * 16 + ch + v;
                int f  = f0 + wn * 32 + ni * 16 + cn_;
                float g  = accg[mi][ni][v] + bup[f];
                float vv = accv[mi][ni][v] + bup[f + DFF_];
                fbf[(size_t)gr * DFF_ + f] = f2bf(__sinf(g) * vv);
            }
}

// Split-K down GEMM: partial[ks] = f[:, ksliсe] @ w_down[:, kslice]^T
// grid (1024/128, 256/64, KSPLIT); K slice = 4096/KSPLIT.
__global__ __launch_bounds__(256) void k_mlp_down(
    const unsigned short* __restrict__ fbf, const unsigned short* __restrict__ dnb,
    float* __restrict__ part) {
    const int tid = threadIdx.x;
    const int wave = tid >> 5, lane = tid & 31;
    const int wm = wave & 1, wn = wave >> 1;
    const int m0 = blockIdx.y * 64;
    const int e0 = blockIdx.x * 128;
    const int kbeg = blockIdx.z * (DFF_ / KSPLIT);
    const int kend = kbeg + (DFF_ / KSPLIT);

    __shared__ __align__(16) unsigned short As[2][64 * LDT];
    __shared__ __align__(16) unsigned short Bs[2][128 * LDT];

    v8f acc[2][2] = {};
    const int arow = tid >> 2, akof = (tid & 3) * 8;
    const int brow = tid >> 1, bkof = (tid & 1) * 16;
    const unsigned aByte = (unsigned)(((size_t)(m0 + arow) * DFF_ + akof) * 2);
    const unsigned bByte = (unsigned)(((size_t)(e0 + brow) * DFF_ + bkof) * 2);
    const int aL = arow * LDT + akof;
    const int bL = brow * LDT + bkof;
    const unsigned ldsA[2] = { lds_addr(&As[0][aL]), lds_addr(&As[1][aL]) };
    const unsigned ldsB[2] = { lds_addr(&Bs[0][bL]), lds_addr(&Bs[1][bL]) };
    const unsigned long long fb  = (unsigned long long)(uintptr_t)fbf;
    const unsigned long long dnb64 = (unsigned long long)(uintptr_t)dnb;

    auto stage = [&](int buf, int k) {
        unsigned kb = (unsigned)k * 2;
        async_b128(ldsA[buf],      aByte + kb,      fb);
        async_b128(ldsB[buf],      bByte + kb,      dnb64);
        async_b128(ldsB[buf] + 16, bByte + kb + 16, dnb64);
    };

    stage(0, kbeg);
    wait_async_all();
    __syncthreads();

    auto step = [&](int cur, int nxt, int kn) {
        if (kn < kend) stage(nxt, kn);
        v16bf a[2], b[2];
#pragma unroll
        for (int mi = 0; mi < 2; ++mi) a[mi] = load_a_frag(As[cur], wm * 32 + mi * 16, lane);
#pragma unroll
        for (int ni = 0; ni < 2; ++ni) b[ni] = load_b_frag(Bs[cur], wn * 32 + ni * 16, lane);
#pragma unroll
        for (int mi = 0; mi < 2; ++mi)
#pragma unroll
            for (int ni = 0; ni < 2; ++ni)
                acc[mi][ni] = __builtin_amdgcn_wmma_f32_16x16x32_bf16(
                    false, a[mi], false, b[ni], (short)0, acc[mi][ni], false, false);
        wait_async_all();
        __syncthreads();
    };

    for (int k0 = kbeg; k0 < kend; k0 += 64) {
        step(0, 1, k0 + 32);
        step(1, 0, k0 + 64);
    }

    float* pslice = part + (size_t)blockIdx.z * ROWS * E_;
    const int ch = (lane >> 4) * 8, cn_ = lane & 15;
#pragma unroll
    for (int mi = 0; mi < 2; ++mi)
#pragma unroll
        for (int ni = 0; ni < 2; ++ni)
#pragma unroll
            for (int v = 0; v < 8; ++v) {
                int gr = m0 + wm * 32 + mi * 16 + ch + v;
                int e  = e0 + wn * 32 + ni * 16 + cn_;
                pslice[(size_t)gr * E_ + e] = acc[mi][ni][v];
            }
}

// Reduce split-K partials + bias + residual -> proc and d_out rows (t<32).
__global__ void k_fix(const float* __restrict__ part, const float* __restrict__ bdn,
                      const float* __restrict__ res2, float* __restrict__ proc,
                      float* __restrict__ out, int chunk) {
    int idx = blockIdx.x * 256 + threadIdx.x;      // over ROWS*E_ = 262144
    int gr = idx >> 10;          // / E_
    int e  = idx & (E_ - 1);
    float o = part[idx] + part[idx + ROWS * E_] + part[idx + 2 * ROWS * E_]
            + part[idx + 3 * ROWS * E_] + bdn[e] + res2[idx];
    proc[idx] = o;
    int b = gr >> 6, t = gr & 63;
    if (t < 32)
        out[((size_t)b * T_ + chunk * 32 + t) * E_ + e] = o;
}

// ---------------------------------------------------------------------------

extern "C" void kernel_launch(void* const* d_in, const int* in_sizes, int n_in,
                              void* d_out, int out_size, void* d_ws, size_t ws_size,
                              hipStream_t stream) {
    const float* x    = (const float*)d_in[0];
    const float* ln1w = (const float*)d_in[1];
    const float* ln1b = (const float*)d_in[2];
    const float* wf1  = (const float*)d_in[3];  // (48,64)
    const float* wf2  = (const float*)d_in[4];  // (64,48)
    const float* ln2w = (const float*)d_in[5];
    const float* ln2b = (const float*)d_in[6];
    const float* wup  = (const float*)d_in[7];  // (8192,1024)
    const float* bup  = (const float*)d_in[8];
    const float* wdn  = (const float*)d_in[9];  // (1024,4096)
    const float* bdn  = (const float*)d_in[10];
    float* out = (float*)d_out;
    (void)in_sizes; (void)n_in; (void)out_size; (void)ws_size;

    char* p = (char*)d_ws;
    auto alloc = [&](size_t bytes) -> char* {
        char* r = p; p += (bytes + 255) & ~(size_t)255; return r;
    };
    unsigned short* upb = (unsigned short*)alloc((size_t)2 * DFF_ * E_ * 2);  // 16 MB
    unsigned short* dnb = (unsigned short*)alloc((size_t)E_ * DFF_ * 2);      //  8 MB
    float* Mmat   = (float*)alloc(64 * 64 * 4);
    float* rowsum = (float*)alloc(64 * 4);
    float* proc0  = (float*)alloc((size_t)ROWS * E_ * 4);                     //  1 MB
    float* proc1  = (float*)alloc((size_t)ROWS * E_ * 4);
    float* res2   = (float*)alloc((size_t)ROWS * E_ * 4);
    unsigned short* cn2 = (unsigned short*)alloc((size_t)ROWS * E_ * 2);      // .5 MB
    unsigned short* fbf = (unsigned short*)alloc((size_t)ROWS * DFF_ * 2);    //  2 MB
    float* part = (float*)alloc((size_t)KSPLIT * ROWS * E_ * 4);              //  4 MB
    float* mu1 = (float*)alloc(ROWS * 4);
    float* rs1 = (float*)alloc(ROWS * 4);

    k_cvt <<<1024, 256, 0, stream>>>(wup, wdn, upb, dnb);
    k_mmat<<<1,    256, 0, stream>>>(wf1, wf2, Mmat, rowsum);
    k_init<<<256,  256, 0, stream>>>(x, proc0);

    float* prev = proc0;
    float* cur  = proc1;
    for (int c = 0; c < NCHUNK; ++c) {
        k_stats1<<<ROWS, 256, 0, stream>>>(prev, x, c, mu1, rs1);
        k_mix   <<<64,   256, 0, stream>>>(prev, x, Mmat, rowsum, mu1, rs1,
                                           ln1w, ln1b, c, res2);
        k_norm2 <<<ROWS, 256, 0, stream>>>(res2, ln2w, ln2b, cn2);
        k_mlp_up  <<<dim3(DFF_ / 128, ROWS / 64), 256, 0, stream>>>(
            cn2, upb, bup, fbf);
        k_mlp_down<<<dim3(E_ / 128, ROWS / 64, KSPLIT), 256, 0, stream>>>(
            fbf, dnb, part);
        k_fix   <<<ROWS * E_ / 256, 256, 0, stream>>>(part, bdn, res2, cur, out, c);
        float* tmp = prev; prev = cur; cur = tmp;
    }
}
